// RelativisticAttention_53618371723768
// MI455X (gfx1250) — matrix operand order repaired and checked by reading, not compiled, not run on hardware
//
#include <hip/hip_runtime.h>

#define NP 8192
#define DM 512

typedef __attribute__((ext_vector_type(8)))  __bf16 v8bf;
typedef __attribute__((ext_vector_type(16))) __bf16 v16bf;
typedef __attribute__((ext_vector_type(8)))  float  v8f;

union BfU { v16bf v; v8bf h[2]; };

__device__ __forceinline__ unsigned short f2bf(float f) {
  unsigned int u = __float_as_uint(f);
  u += 0x7FFFu + ((u >> 16) & 1u);   // round-to-nearest-even
  return (unsigned short)(u >> 16);
}

// Load a 16-element bf16 operand as two 8-element chunks: p[0..7], p[stride..stride+7].
// A-operand as loaded here: lane m = row m%16, K = hi*16 .. hi*16+15 -> stride 8
// B-operand layout: lane n = col n%16, K = hi*16 .. hi*16+15          -> stride 8
__device__ __forceinline__ v16bf load_pair(const unsigned short* p, int stride) {
  BfU u;
  u.h[0] = *(const v8bf*)(p);
  u.h[1] = *(const v8bf*)(p + stride);
  return u.v;
}

__device__ __forceinline__ float rmax16(float x) {
#pragma unroll
  for (int m = 1; m < 16; m <<= 1) x = fmaxf(x, __shfl_xor(x, m, 32));
  return x;
}
__device__ __forceinline__ float rsum16(float x) {
#pragma unroll
  for (int m = 1; m < 16; m <<= 1) x += __shfl_xor(x, m, 32);
  return x;
}

// ---------------- Kernel 0: f32 -> bf16 convert ----------------
__global__ void cvt_kernel(const float* __restrict__ src,
                           unsigned short* __restrict__ dst, int n) {
  int i = blockIdx.x * blockDim.x + threadIdx.x;
  if (i < n) dst[i] = f2bf(src[i]);
}

// ---------------- Kernel 3: light-cone mask -> d_out[N*D ..] ----------------
__global__ void mask_kernel(const float* __restrict__ coords,
                            float* __restrict__ mk) {
  size_t idx = (size_t)blockIdx.x * blockDim.x + threadIdx.x;
  int i = (int)(idx >> 13);          // NP == 2^13
  int j = (int)(idx & (NP - 1));
  float4 ci = *(const float4*)(coords + (size_t)i * 4);
  float4 cj = *(const float4*)(coords + (size_t)j * 4);
  float dt = ci.x - cj.x, dx = ci.y - cj.y, dy = ci.z - cj.z, dz = ci.w - cj.w;
  float ds2 = dt * dt - dx * dx - dy * dy - dz * dz;
  mk[idx] = ((ds2 >= 0.0f) && (dt >= 0.0f)) ? 1.0f : 0.0f;
}

// ---------------- Kernel 1: projections via WMMA ----------------
// z=0: Q = (F*Wq^T + bq) * (1/sqrt(D))   [NP x DM], row-major bf16
// z=1: K =  F*Wk^T + bk                  [NP x DM], row-major bf16
// z=2: Vt = Wv*F^T + bv (per dim row)    [DM x NP], row-major bf16 (V transposed)
__global__ __launch_bounds__(256) void proj_kernel(
    const unsigned short* __restrict__ Fb,
    const unsigned short* __restrict__ Wqb,
    const unsigned short* __restrict__ Wkb,
    const unsigned short* __restrict__ Wvb,
    const float* __restrict__ bq, const float* __restrict__ bk,
    const float* __restrict__ bv,
    unsigned short* __restrict__ Qb, unsigned short* __restrict__ Kb,
    unsigned short* __restrict__ Vt) {
  const int lane = threadIdx.x & 31;
  const int wv   = threadIdx.x >> 5;
  const int hi   = lane >> 4;
  const int ln   = lane & 15;
  const int z    = blockIdx.y;
  const int tid  = blockIdx.x * 8 + wv;   // 16384 16x16 tiles per matrix

  const unsigned short* Ab;
  const unsigned short* Bb;
  int rowT, colT;
  if (z < 2) { rowT = tid >> 5; colT = tid & 31; Ab = Fb;  Bb = (z == 0) ? Wqb : Wkb; }
  else       { rowT = tid & 31; colT = tid >> 5; Ab = Wvb; Bb = Fb; }

  const unsigned short* ap = Ab + (size_t)(rowT * 16 + ln) * DM + hi * 16;
  const unsigned short* bp = Bb + (size_t)(colT * 16 + ln) * DM + hi * 16;

  v8f acc = {};
  for (int kk = 0; kk < DM; kk += 32) {
    v16bf a = load_pair(ap + kk, 8);
    v16bf b = load_pair(bp + kk, 8);
    acc = __builtin_amdgcn_wmma_f32_16x16x32_bf16(false, a, false, b,
                                                  (short)0, acc, false, false);
  }

  if (z < 2) {
    const float* bias = (z == 0) ? bq : bk;
    unsigned short* dst = (z == 0) ? Qb : Kb;
    const float scale = (z == 0) ? 0.044194173824159216f : 1.0f; // 1/sqrt(512)
    float bcol = bias[colT * 16 + ln];
#pragma unroll
    for (int c = 0; c < 8; ++c) {
      int row = rowT * 16 + c + 8 * hi;
      dst[(size_t)row * DM + colT * 16 + ln] = f2bf((acc[c] + bcol) * scale);
    }
  } else {
#pragma unroll
    for (int c = 0; c < 8; ++c) {
      int dim = rowT * 16 + c + 8 * hi;
      Vt[(size_t)dim * NP + colT * 16 + ln] = f2bf(acc[c] + bv[dim]);
    }
  }
}

// ---------------- Kernel 2: flash attention ----------------
// Workgroup = 8 waves over 32 query rows:
//   waves 0-3 -> rows [qb, qb+16), waves 4-7 -> rows [qb+16, qb+32)
//   wave (w&3) owns output dims [128*(w&3), 128*(w&3)+128)  -> 4x score redundancy
// Waves are fully independent (no cross-wave barriers).
__global__ __launch_bounds__(256) void attn_kernel(
    const float* __restrict__ coords,
    const unsigned short* __restrict__ Qb,
    const unsigned short* __restrict__ Kb,
    const unsigned short* __restrict__ Vt,
    float* __restrict__ out) {
  __shared__ unsigned short ldsP[8][16 * 32];   // per-wave P staging (bf16)

  const int lane = threadIdx.x & 31;
  const int wv   = threadIdx.x >> 5;
  const int hi   = lane >> 4;
  const int ln   = lane & 15;
  const int qb   = blockIdx.x * 32 + (wv >> 2) * 16;
  const int db   = (wv & 3) * 128;

  // C-layout row coords: component c <-> row qb + c + 8*hi
  float cit[8], cix[8], ciy[8], ciz[8];
#pragma unroll
  for (int c = 0; c < 8; ++c) {
    const float4 cc = *(const float4*)(coords + (size_t)(qb + c + 8 * hi) * 4);
    cit[c] = cc.x; cix[c] = cc.y; ciy[c] = cc.z; ciz[c] = cc.w;
  }

  float m[8], l[8];
  v8f o[8];
#pragma unroll
  for (int t = 0; t < 8; ++t) o[t] = (v8f){};
#pragma unroll
  for (int c = 0; c < 8; ++c) { m[c] = -__builtin_inff(); l[c] = 0.0f; }

  const unsigned short* qp = Qb + (size_t)(qb + ln) * DM + hi * 16;
  unsigned short* pp = ldsP[wv];

  for (int kb = 0; kb < NP; kb += 32) {
    const unsigned short* kp0 = Kb + (size_t)(kb + ln) * DM + hi * 16;
    const unsigned short* kp1 = kp0 + (size_t)16 * DM;

    // prefetch next iteration's K rows and V columns (global_prefetch_b8);
    // whole K/V working set sits in L2 (192 MB), this hides L2->L0 latency.
    if (kb + 32 < NP) {
      __builtin_prefetch(kp0 + (size_t)32 * DM + hi * 256, 0, 1);
      __builtin_prefetch(kp1 + (size_t)32 * DM + hi * 256, 0, 1);
      __builtin_prefetch(Vt + (size_t)(db + lane) * NP + kb + 32, 0, 1);
      __builtin_prefetch(Vt + (size_t)(db + 64 + lane) * NP + kb + 32, 0, 1);
    }

    // ---- scores for 32 keys: two 16x16 tiles, K-reduce over D=512 ----
    v8f s0 = {}, s1 = {};
    for (int kk = 0; kk < DM; kk += 32) {
      v16bf a  = load_pair(qp + kk, 8);
      v16bf b0 = load_pair(kp0 + kk, 8);
      v16bf b1 = load_pair(kp1 + kk, 8);
      s0 = __builtin_amdgcn_wmma_f32_16x16x32_bf16(false, a, false, b0,
                                                   (short)0, s0, false, false);
      s1 = __builtin_amdgcn_wmma_f32_16x16x32_bf16(false, a, false, b1,
                                                   (short)0, s1, false, false);
    }

    // ---- mask + online softmax (rows live in lane halves) ----
    const float4 cj0 = *(const float4*)(coords + (size_t)(kb + ln) * 4);
    const float4 cj1 = *(const float4*)(coords + (size_t)(kb + 16 + ln) * 4);
    float alpha[8];
#pragma unroll
    for (int c = 0; c < 8; ++c) {
      float dt0 = cit[c] - cj0.x, dx0 = cix[c] - cj0.y, dy0 = ciy[c] - cj0.z, dz0 = ciz[c] - cj0.w;
      float dt1 = cit[c] - cj1.x, dx1 = cix[c] - cj1.y, dy1 = ciy[c] - cj1.z, dz1 = ciz[c] - cj1.w;
      bool ok0 = ((dt0 * dt0 - dx0 * dx0 - dy0 * dy0 - dz0 * dz0) >= 0.0f) && (dt0 >= 0.0f);
      bool ok1 = ((dt1 * dt1 - dx1 * dx1 - dy1 * dy1 - dz1 * dz1) >= 0.0f) && (dt1 >= 0.0f);
      float sv0 = ok0 ? s0[c] : -__builtin_inff();
      float sv1 = ok1 ? s1[c] : -__builtin_inff();
      float mn = fmaxf(m[c], rmax16(fmaxf(sv0, sv1)));
      alpha[c] = (mn == m[c]) ? 1.0f : __expf(m[c] - mn);  // handles -inf==-inf
      float p0 = ok0 ? __expf(sv0 - mn) : 0.0f;
      float p1 = ok1 ? __expf(sv1 - mn) : 0.0f;
      l[c] = l[c] * alpha[c] + rsum16(p0 + p1);
      m[c] = mn;
      pp[(c + 8 * hi) * 32 + ln]      = f2bf(p0);   // P row-major 16x32 in LDS
      pp[(c + 8 * hi) * 32 + 16 + ln] = f2bf(p1);
    }
#pragma unroll
    for (int t = 0; t < 8; ++t) {
#pragma unroll
      for (int c = 0; c < 8; ++c) o[t][c] *= alpha[c];
    }

    // C-layout -> A-layout round trip through LDS (wave-private, DS in-order)
    asm volatile("s_wait_dscnt 0" ::: "memory");
    v16bf pA = load_pair(pp + ln * 32 + hi * 16, 8);

    // ---- O += P(16x32) * V(32x128), V stored transposed -> contiguous B loads ----
#pragma unroll
    for (int t = 0; t < 8; ++t) {
      v16bf bv = load_pair(Vt + (size_t)(db + t * 16 + ln) * NP + kb + hi * 16, 8);
      o[t] = __builtin_amdgcn_wmma_f32_16x16x32_bf16(false, pA, false, bv,
                                                     (short)0, o[t], false, false);
    }
  }

  // ---- epilogue: O / l (fully-masked rows -> 0, matches nan_to_num) ----
#pragma unroll
  for (int c = 0; c < 8; ++c) {
    float inv = (l[c] > 0.0f) ? (1.0f / l[c]) : 0.0f;
    size_t row = (size_t)(qb + c + 8 * hi) * DM;
#pragma unroll
    for (int t = 0; t < 8; ++t) out[row + db + t * 16 + ln] = o[t][c] * inv;
  }
}

extern "C" void kernel_launch(void* const* d_in, const int* in_sizes, int n_in,
                              void* d_out, int out_size, void* d_ws, size_t ws_size,
                              hipStream_t stream) {
  const float* coords   = (const float*)d_in[0];
  const float* features = (const float*)d_in[1];
  const float* Wq = (const float*)d_in[2];
  const float* bq = (const float*)d_in[3];
  const float* Wk = (const float*)d_in[4];
  const float* bk = (const float*)d_in[5];
  const float* Wv = (const float*)d_in[6];
  const float* bv = (const float*)d_in[7];
  float* out = (float*)d_out;

  // workspace layout (bf16 buffers)
  unsigned char* ws = (unsigned char*)d_ws;
  const size_t SZ_F = (size_t)NP * DM * 2;   // 8 MB
  const size_t SZ_W = (size_t)DM * DM * 2;   // 0.5 MB
  unsigned short* Fb  = (unsigned short*)(ws);
  unsigned short* Wqb = (unsigned short*)(ws + SZ_F);
  unsigned short* Wkb = (unsigned short*)(ws + SZ_F + SZ_W);
  unsigned short* Wvb = (unsigned short*)(ws + SZ_F + 2 * SZ_W);
  unsigned short* Qb  = (unsigned short*)(ws + SZ_F + 3 * SZ_W);
  unsigned short* Kb  = (unsigned short*)(ws + 2 * SZ_F + 3 * SZ_W);
  unsigned short* Vtb = (unsigned short*)(ws + 3 * SZ_F + 3 * SZ_W);

  cvt_kernel<<<dim3((NP * DM) / 256), dim3(256), 0, stream>>>(features, Fb, NP * DM);
  cvt_kernel<<<dim3((DM * DM) / 256), dim3(256), 0, stream>>>(Wq, Wqb, DM * DM);
  cvt_kernel<<<dim3((DM * DM) / 256), dim3(256), 0, stream>>>(Wk, Wkb, DM * DM);
  cvt_kernel<<<dim3((DM * DM) / 256), dim3(256), 0, stream>>>(Wv, Wvb, DM * DM);

  proj_kernel<<<dim3(2048, 3), dim3(256), 0, stream>>>(
      Fb, Wqb, Wkb, Wvb, bq, bk, bv, Qb, Kb, Vtb);

  mask_kernel<<<dim3(((size_t)NP * NP) / 256), dim3(256), 0, stream>>>(
      coords, out + (size_t)NP * DM);

  attn_kernel<<<dim3(NP / 32), dim3(256), 0, stream>>>(coords, Qb, Kb, Vtb, out);
}
